// SparseEdgeFullLayer_17549236371611
// MI455X (gfx1250) — matrix operand order repaired and checked by reading, not compiled, not run on hardware
//
#include <hip/hip_runtime.h>

// ---------------- problem constants ----------------
constexpr int kN = 50000;   // nodes
constexpr int kE = 640000;  // edges
constexpr int kD = 128;     // model dim
constexpr int kH = 8;       // heads
// C = 16 per head; scale = 1/sqrt(16) = 0.25

typedef __attribute__((ext_vector_type(16))) __bf16 bf16x16;
typedef __attribute__((ext_vector_type(8)))  float  f32x8;

#define WMMA_BF16(C, A, B) \
    C = __builtin_amdgcn_wmma_f32_16x16x32_bf16(false, A, false, B, (short)0, C, false, false)

// ---------------- CDNA5 async global->LDS copy (16B per call) ----------------
__device__ __forceinline__ void async_copy16(void* lds_dst, const void* gsrc) {
    unsigned l = (unsigned)(uintptr_t)lds_dst;            // low 32 bits = LDS offset
    unsigned long long g = (unsigned long long)(uintptr_t)gsrc;
    asm volatile("global_load_async_to_lds_b128 %0, %1, off"
                 :: "v"(l), "v"(g) : "memory");
}
__device__ __forceinline__ void async_wait0() {
    asm volatile("s_wait_asynccnt 0x0" ::: "memory");
}

// ---------------- fragment layout helpers ----------------
// A-frag buffer layout: afrag[(ktile*32 + lane)*16 + e]  (bf16)
// lane = row + 16*kh ; e = 8*(koff>=16) + (koff&7) ; kh = (koff>>3)&1 ; koff = k&31
__device__ __forceinline__ bf16x16 load_a_frag(const __bf16* afrag, int ktile, int lane) {
    return *(const bf16x16*)(afrag + ((ktile * 32 + lane) << 4));
}
// B packed weights: Wpk[((coltile*nkt + ktile)*32 + lane)*16 + e]
__device__ __forceinline__ bf16x16 load_b_pk(const __bf16* Wpk, int tileidx, int lane) {
    return *(const bf16x16*)(Wpk + (((size_t)tileidx * 32 + lane) << 4));
}
// write 8 consecutive-K f32 values (c0 multiple of 8) into A-frag buffer (one b128 store)
__device__ __forceinline__ void store_a_frag8(__bf16* afrag, int row, int c0, const float* v8) {
    int ktile = c0 >> 5, koff = c0 & 31;
    int kh   = (koff >> 3) & 1;
    int lane = row + (kh << 4);
    int e    = (koff & 16) >> 1;                  // 0 or 8
    __bf16* d = afrag + ((ktile * 32 + lane) << 4) + e;
#pragma unroll
    for (int j = 0; j < 8; ++j) d[j] = (__bf16)v8[j];
}
// scatter one C fragment (rows m, col = next-GEMM K) into an A-frag buffer, w/ bias(+relu)
__device__ __forceinline__ void store_c_to_afrag(__bf16* afrag, const f32x8& c, float bias,
                                                 int colbase, int lane) {
    int k = colbase + (lane & 15);
    int ktile = k >> 5, koff = k & 31;
    int kh = (koff >> 3) & 1;
    int e  = ((koff & 16) >> 1) + (koff & 7);
    int mb = (lane >> 4) << 3;
#pragma unroll
    for (int i = 0; i < 8; ++i) {
        int laneA = (mb + i) + (kh << 4);
        afrag[((ktile * 32 + laneA) << 4) + e] = (__bf16)fmaxf(c[i] + bias, 0.0f);
    }
}

// ---------------- KP: pack f32 weight [K][Ncols] -> bf16 B-fragment order ----------------
__global__ void k_pack(const float* __restrict__ W, int K, int Ncols, __bf16* __restrict__ Wpk) {
    int i = blockIdx.x * 256 + threadIdx.x;          // i = (coltile*nkt + ktile)*32 + lane
    int nkt = K >> 5;
    int total = (Ncols >> 4) * nkt * 32;
    if (i >= total) return;
    int lane  = i & 31;
    int ktile = (i >> 5) % nkt;
    int coltile = (i >> 5) / nkt;
    int col = coltile * 16 + (lane & 15);
    int kb  = ktile * 32 + ((lane >> 4) << 3);
    __bf16* d = Wpk + ((size_t)i << 4);
#pragma unroll
    for (int e = 0; e < 16; ++e) d[e] = (__bf16)W[(size_t)(kb + e + ((e >> 3) << 3)) * Ncols + col];
}

// ---------------- K0: init segment max (ordered-uint -inf) and Z ----------------
__global__ void k0_init(unsigned* __restrict__ m_u, float* __restrict__ Z) {
    int i = blockIdx.x * 256 + threadIdx.x;
    if (i < kN * kH) { m_u[i] = 0u; Z[i] = 0.0f; }
}

// ---------------- K1: q = x@Wq+bq ; acc = x@Wskip+bskip ----------------
__global__ void __launch_bounds__(256) k1_node_proj(
        const float* __restrict__ x,
        const __bf16* __restrict__ Wq_pk, const float* __restrict__ bq,
        const __bf16* __restrict__ Ws_pk, const float* __restrict__ bs,
        float* __restrict__ q_node, float* __restrict__ acc) {
    __shared__ __attribute__((aligned(16))) float  xraw[16 * 128];
    __shared__ __attribute__((aligned(32))) __bf16 afrag[4 * 32 * 16];
    const int t = threadIdx.x;
    const int row0 = blockIdx.x * 16;
    {   // async-stage the contiguous 8KB x-tile into LDS
        const char* g = (const char*)(x + (size_t)row0 * kD);
        char* l = (char*)xraw;
        async_copy16(l + t * 16,        g + t * 16);
        async_copy16(l + 4096 + t * 16, g + 4096 + t * 16);
        async_wait0();
    }
    __syncthreads();
    {   // build bf16 A fragments
        int row = t >> 4, c0 = (t & 15) * 8;
        store_a_frag8(afrag, row, c0, xraw + row * kD + c0);
    }
    __syncthreads();
    const int wave = t >> 5, lane = t & 31;
    f32x8 cq = {}, cs = {};
#pragma unroll
    for (int kt = 0; kt < 4; ++kt) {
        bf16x16 a  = load_a_frag(afrag, kt, lane);
        bf16x16 b0 = load_b_pk(Wq_pk, wave * 4 + kt, lane);
        WMMA_BF16(cq, a, b0);
        bf16x16 b1 = load_b_pk(Ws_pk, wave * 4 + kt, lane);
        WMMA_BF16(cs, a, b1);
    }
    const int col = wave * 16 + (lane & 15);
    const int mb  = (lane >> 4) << 3;
    const float bqv = bq[col], bsv = bs[col];
#pragma unroll
    for (int i = 0; i < 8; ++i) {
        size_t r = (size_t)(row0 + mb + i) * kD + col;
        q_node[r] = cq[i] + bqv;
        acc[r]    = cs[i] + bsv;
    }
}

// ---------------- K2: per-edge k,v GEMMs, alpha, segment max ----------------
__global__ void __launch_bounds__(256) k2_edge(
        const float* __restrict__ x, const int* __restrict__ ei,
        const float* __restrict__ ea,
        const __bf16* __restrict__ Wk_pk, const float* __restrict__ bk,
        const __bf16* __restrict__ Wv_pk, const float* __restrict__ bv,
        const float* __restrict__ q_node,
        float* __restrict__ alpha, __bf16* __restrict__ v_edge,
        unsigned* __restrict__ m_u) {
    __shared__ __attribute__((aligned(32))) __bf16 afrag[4 * 32 * 16];
    __shared__ __attribute__((aligned(16))) float  kt_[16 * 128];
    __shared__ __attribute__((aligned(16))) __bf16 vt[16 * 128];
    __shared__ int sdst[16];
    const int e0 = blockIdx.x * 16;
    const int t  = threadIdx.x;
    if (t < 16) sdst[t] = ei[kE + e0 + t];
    {   // kv_in = x[src] + edge_attr -> bf16 A fragments
        const int row = t >> 4, c0 = (t & 15) * 8;
        const int src = ei[e0 + row];
        const float4* xa = (const float4*)(x + (size_t)src * kD + c0);
        const float4* eb = (const float4*)(ea + (size_t)(e0 + row) * kD + c0);
        float4 a0 = xa[0], a1 = xa[1], b0 = eb[0], b1 = eb[1];
        float s8[8] = {a0.x + b0.x, a0.y + b0.y, a0.z + b0.z, a0.w + b0.w,
                       a1.x + b1.x, a1.y + b1.y, a1.z + b1.z, a1.w + b1.w};
        store_a_frag8(afrag, row, c0, s8);
    }
    __syncthreads();
    const int wave = t >> 5, lane = t & 31;
    f32x8 ck = {}, cv = {};
#pragma unroll
    for (int kt = 0; kt < 4; ++kt) {
        bf16x16 a  = load_a_frag(afrag, kt, lane);
        bf16x16 bK = load_b_pk(Wk_pk, wave * 4 + kt, lane);
        WMMA_BF16(ck, a, bK);
        bf16x16 bV = load_b_pk(Wv_pk, wave * 4 + kt, lane);
        WMMA_BF16(cv, a, bV);
    }
    const int col = wave * 16 + (lane & 15);
    const int mb  = (lane >> 4) << 3;
    const float bkv = bk[col], bvv = bv[col];
#pragma unroll
    for (int i = 0; i < 8; ++i) {
        int m = mb + i;
        kt_[m * kD + col] = ck[i] + bkv;
        vt[m * kD + col]  = (__bf16)(cv[i] + bvv);
    }
    __syncthreads();
    {   // coalesced bf16 v-tile flush: one b128 store per thread
        const uint4* s = (const uint4*)vt;
        uint4* d = (uint4*)(v_edge + (size_t)e0 * kD);
        d[t] = s[t];
    }
    // alpha[e,h] = dot16(q[dst[e],h], k[e,h]) * 0.25 ; head h == column tile h
    if (t < 128) {
        const int e = t >> 3, h = t & 7;
        const int d = sdst[e];
        const float4* qp = (const float4*)(q_node + (size_t)d * kD + h * 16);
        const float4* kp = (const float4*)(kt_ + e * kD + h * 16);
        float s = 0.f;
#pragma unroll
        for (int c = 0; c < 4; ++c) {
            float4 q4 = qp[c], k4 = kp[c];
            s += q4.x * k4.x + q4.y * k4.y + q4.z * k4.z + q4.w * k4.w;
        }
        s *= 0.25f;
        alpha[(size_t)(e0 + e) * kH + h] = s;
        unsigned enc = __float_as_uint(s) ^
                       ((unsigned)(__float_as_int(s) >> 31) | 0x80000000u);
        atomicMax(m_u + (size_t)d * kH + h, enc);
    }
}

// ---------------- K3: ex = exp(alpha - m[dst]) ; Z[dst] += ex ----------------
__global__ void k3_expsum(const int* __restrict__ ei, float* __restrict__ alpha,
                          const unsigned* __restrict__ m_u, float* __restrict__ Z) {
    size_t i = (size_t)blockIdx.x * 256 + threadIdx.x;
    if (i >= (size_t)kE * kH) return;
    size_t e = i >> 3; int h = (int)(i & 7);
    int d = ei[kE + e];
    unsigned u = m_u[(size_t)d * kH + h];
    float mv = (u & 0x80000000u) ? __uint_as_float(u ^ 0x80000000u)
                                 : __uint_as_float(~u);
    float ex = __expf(alpha[i] - mv);
    alpha[i] = ex;
    atomicAdd(Z + (size_t)d * kH + h, ex);
}

// ---------------- K4: acc[dst] += v * ex/Z (weighted scatter) ----------------
__global__ void k4_scatter(const int* __restrict__ ei, const float* __restrict__ alpha,
                           const float* __restrict__ Z, const __bf16* __restrict__ v_edge,
                           float* __restrict__ acc) {
    size_t i = (size_t)blockIdx.x * 256 + threadIdx.x;   // over E*32
    size_t e = i >> 5;
    int q  = (int)(i & 31);
    int d0 = q * 4, h = d0 >> 4;
    int dn = ei[kE + e];
    float w = alpha[e * kH + h] / (Z[(size_t)dn * kH + h] + 1e-16f);
    uint2 vv = *(const uint2*)(v_edge + e * (size_t)kD + d0);
    float* op = acc + (size_t)dn * kD + d0;
    atomicAdd(op + 0, __uint_as_float(vv.x << 16)          * w);
    atomicAdd(op + 1, __uint_as_float(vv.x & 0xFFFF0000u)  * w);
    atomicAdd(op + 2, __uint_as_float(vv.y << 16)          * w);
    atomicAdd(op + 3, __uint_as_float(vv.y & 0xFFFF0000u)  * w);
}

// ---------------- K5: h1 = LN1(x + acc@WO + bO) ----------------
__global__ void __launch_bounds__(256) k5_outproj_ln1(
        const float* __restrict__ x, const float* __restrict__ acc,
        const __bf16* __restrict__ WO_pk, const float* __restrict__ bO,
        const float* __restrict__ g, const float* __restrict__ b,
        float* __restrict__ h1) {
    __shared__ __attribute__((aligned(16))) float  araw[16 * 128];
    __shared__ __attribute__((aligned(16))) float  xraw[16 * 128];
    __shared__ __attribute__((aligned(32))) __bf16 afrag[4 * 32 * 16];
    __shared__ float ot[16 * 128];
    __shared__ float mu[16], rs[16];
    const int t = threadIdx.x;
    const int row0 = blockIdx.x * 16;
    {   // async-stage both contiguous tiles
        const char* ga = (const char*)(acc + (size_t)row0 * kD);
        const char* gx = (const char*)(x + (size_t)row0 * kD);
        async_copy16((char*)araw + t * 16,        ga + t * 16);
        async_copy16((char*)araw + 4096 + t * 16, ga + 4096 + t * 16);
        async_copy16((char*)xraw + t * 16,        gx + t * 16);
        async_copy16((char*)xraw + 4096 + t * 16, gx + 4096 + t * 16);
        async_wait0();
    }
    __syncthreads();
    {
        int row = t >> 4, c0 = (t & 15) * 8;
        store_a_frag8(afrag, row, c0, araw + row * kD + c0);
    }
    __syncthreads();
    const int wave = t >> 5, lane = t & 31;
    f32x8 c = {};
#pragma unroll
    for (int kt = 0; kt < 4; ++kt) {
        bf16x16 a  = load_a_frag(afrag, kt, lane);
        bf16x16 bW = load_b_pk(WO_pk, wave * 4 + kt, lane);
        WMMA_BF16(c, a, bW);
    }
    const int col = wave * 16 + (lane & 15);
    const int mb  = (lane >> 4) << 3;
    const float bo = bO[col];
#pragma unroll
    for (int i = 0; i < 8; ++i) {
        int m = mb + i;
        ot[m * kD + col] = c[i] + bo + xraw[m * kD + col];
    }
    __syncthreads();
    if (t < 16) {
        const float* p = ot + t * kD;
        float s = 0.f;
        for (int j = 0; j < kD; ++j) s += p[j];
        float m = s * (1.0f / kD), v = 0.f;
        for (int j = 0; j < kD; ++j) { float dd = p[j] - m; v += dd * dd; }
        mu[t] = m; rs[t] = rsqrtf(v * (1.0f / kD) + 1e-5f);
    }
    __syncthreads();
    const int row = t >> 4, c0 = (t & 15) * 8;
    const float m = mu[row], r = rs[row];
    float* op = h1 + (size_t)(row0 + row) * kD + c0;
    const float* ip = ot + row * kD + c0;
#pragma unroll
    for (int j = 0; j < 8; ++j) op[j] = (ip[j] - m) * r * g[c0 + j] + b[c0 + j];
}

// ---------------- K6: out = LN2(h1 + relu(h1@W1+b1)@W2+b2) ----------------
__global__ void __launch_bounds__(256) k6_ffn_ln2(
        const float* __restrict__ h1,
        const __bf16* __restrict__ W1_pk, const float* __restrict__ b1,
        const __bf16* __restrict__ W2_pk, const float* __restrict__ b2,
        const float* __restrict__ g, const float* __restrict__ b,
        float* __restrict__ out) {
    __shared__ __attribute__((aligned(16))) float  hraw[16 * 128];
    __shared__ __attribute__((aligned(32))) __bf16 afrag1[4 * 32 * 16];
    __shared__ __attribute__((aligned(32))) __bf16 afrag2[8 * 32 * 16];
    __shared__ float ot[16 * 128];
    __shared__ float mu[16], rs[16];
    const int t = threadIdx.x;
    const int row0 = blockIdx.x * 16;
    {
        const char* gh = (const char*)(h1 + (size_t)row0 * kD);
        async_copy16((char*)hraw + t * 16,        gh + t * 16);
        async_copy16((char*)hraw + 4096 + t * 16, gh + 4096 + t * 16);
        async_wait0();
    }
    __syncthreads();
    {
        int row = t >> 4, c0 = (t & 15) * 8;
        store_a_frag8(afrag1, row, c0, hraw + row * kD + c0);
    }
    __syncthreads();
    const int wave = t >> 5, lane = t & 31;
    // GEMM1: [16x128]@[128x256] + b1, relu -> afrag2 (A for GEMM2)
#pragma unroll
    for (int ct = wave; ct < 16; ct += 8) {
        f32x8 c = {};
#pragma unroll
        for (int kt = 0; kt < 4; ++kt) {
            bf16x16 a  = load_a_frag(afrag1, kt, lane);
            bf16x16 bW = load_b_pk(W1_pk, ct * 4 + kt, lane);
            WMMA_BF16(c, a, bW);
        }
        store_c_to_afrag(afrag2, c, b1[ct * 16 + (lane & 15)], ct * 16, lane);
    }
    __syncthreads();
    // GEMM2: [16x256]@[256x128] + b2 + h1 -> ot
    {
        f32x8 c = {};
#pragma unroll
        for (int kt = 0; kt < 8; ++kt) {
            bf16x16 a  = load_a_frag(afrag2, kt, lane);
            bf16x16 bW = load_b_pk(W2_pk, wave * 8 + kt, lane);
            WMMA_BF16(c, a, bW);
        }
        const int col = wave * 16 + (lane & 15);
        const int mb  = (lane >> 4) << 3;
        const float bb = b2[col];
#pragma unroll
        for (int i = 0; i < 8; ++i) {
            int m = mb + i;
            ot[m * kD + col] = c[i] + bb + hraw[m * kD + col];
        }
    }
    __syncthreads();
    if (t < 16) {
        const float* p = ot + t * kD;
        float s = 0.f;
        for (int j = 0; j < kD; ++j) s += p[j];
        float m = s * (1.0f / kD), v = 0.f;
        for (int j = 0; j < kD; ++j) { float dd = p[j] - m; v += dd * dd; }
        mu[t] = m; rs[t] = rsqrtf(v * (1.0f / kD) + 1e-5f);
    }
    __syncthreads();
    const int row = t >> 4, c0 = (t & 15) * 8;
    const float m = mu[row], r = rs[row];
    float* op = out + (size_t)(row0 + row) * kD + c0;
    const float* ip = ot + row * kD + c0;
#pragma unroll
    for (int j = 0; j < 8; ++j) op[j] = (ip[j] - m) * r * g[c0 + j] + b[c0 + j];
}

// ---------------- launcher ----------------
extern "C" void kernel_launch(void* const* d_in, const int* in_sizes, int n_in,
                              void* d_out, int out_size, void* d_ws, size_t ws_size,
                              hipStream_t stream) {
    const float* x    = (const float*)d_in[0];
    const int*   ei   = (const int*)  d_in[1];   // [2,E]
    const float* ea   = (const float*)d_in[2];
    const float* Wq   = (const float*)d_in[3];
    const float* bq   = (const float*)d_in[4];
    const float* Wk   = (const float*)d_in[5];
    const float* bk   = (const float*)d_in[6];
    const float* Wv   = (const float*)d_in[7];
    const float* bv   = (const float*)d_in[8];
    const float* Wsk  = (const float*)d_in[9];
    const float* bsk  = (const float*)d_in[10];
    const float* WO   = (const float*)d_in[11];
    const float* bO   = (const float*)d_in[12];
    const float* ln1g = (const float*)d_in[13];
    const float* ln1b = (const float*)d_in[14];
    const float* W1   = (const float*)d_in[15];
    const float* b1   = (const float*)d_in[16];
    const float* W2   = (const float*)d_in[17];
    const float* b2   = (const float*)d_in[18];
    const float* ln2g = (const float*)d_in[19];
    const float* ln2b = (const float*)d_in[20];
    float* out = (float*)d_out;

    // workspace carve-up
    float*    q_node = (float*)d_ws;                           // N*D f32
    float*    acc    = q_node + (size_t)kN * kD;               // N*D f32
    float*    h1     = acc    + (size_t)kN * kD;               // N*D f32
    float*    alpha  = h1     + (size_t)kN * kD;               // E*H f32
    unsigned* m_u    = (unsigned*)(alpha + (size_t)kE * kH);   // N*H u32
    float*    Z      = (float*)(m_u + (size_t)kN * kH);        // N*H f32
    __bf16*   v_edge = (__bf16*)(Z + (size_t)kN * kH);         // E*D bf16
    __bf16*   Wq_pk  = v_edge + (size_t)kE * kD;               // packed weights (bf16)
    __bf16*   Wk_pk  = Wq_pk + kD * kD;
    __bf16*   Wv_pk  = Wk_pk + kD * kD;
    __bf16*   Wsk_pk = Wv_pk + kD * kD;
    __bf16*   WO_pk  = Wsk_pk + kD * kD;
    __bf16*   W1_pk  = WO_pk + kD * kD;                        // 128x256
    __bf16*   W2_pk  = W1_pk + kD * 2 * kD;                    // 256x128

    const int nodeTiles = kN / 16;   // 3125
    const int edgeTiles = kE / 16;   // 40000

    // pack weights (tiny; every call, idempotent)
    k_pack<<<4, 256, 0, stream>>>(Wq,  kD, kD,     Wq_pk);
    k_pack<<<4, 256, 0, stream>>>(Wk,  kD, kD,     Wk_pk);
    k_pack<<<4, 256, 0, stream>>>(Wv,  kD, kD,     Wv_pk);
    k_pack<<<4, 256, 0, stream>>>(Wsk, kD, kD,     Wsk_pk);
    k_pack<<<4, 256, 0, stream>>>(WO,  kD, kD,     WO_pk);
    k_pack<<<8, 256, 0, stream>>>(W1,  kD, 2 * kD, W1_pk);
    k_pack<<<8, 256, 0, stream>>>(W2,  2 * kD, kD, W2_pk);

    k0_init<<<(kN * kH + 255) / 256, 256, 0, stream>>>(m_u, Z);
    k1_node_proj<<<nodeTiles, 256, 0, stream>>>(x, Wq_pk, bq, Wsk_pk, bsk, q_node, acc);
    k2_edge<<<edgeTiles, 256, 0, stream>>>(x, ei, ea, Wk_pk, bk, Wv_pk, bv, q_node,
                                           alpha, v_edge, m_u);
    k3_expsum<<<(kE * kH) / 256, 256, 0, stream>>>(ei, alpha, m_u, Z);
    k4_scatter<<<(kE * 32) / 256, 256, 0, stream>>>(ei, alpha, Z, v_edge, acc);
    k5_outproj_ln1<<<nodeTiles, 256, 0, stream>>>(x, acc, WO_pk, bO, ln1g, ln1b, h1);
    k6_ffn_ln2<<<nodeTiles, 256, 0, stream>>>(h1, W1_pk, b1, W2_pk, b2, ln2g, ln2b, out);
}
